// TransformerNetwork32_dip_v2_18124761989683
// MI455X (gfx1250) — matrix operand order repaired and checked by reading, not compiled, not run on hardware
//
#include <hip/hip_runtime.h>
#include <hip/hip_bf16.h>

// ---------------------------------------------------------------------------
// One workgroup per patch (r = 1024). 640 threads = 20 wave32s; each wave owns
// one 16-row query block of the L=320 sequence and runs flash-attention with
// V_WMMA_F32_16X16X32_F16 for both Q*K^T and P*V (K padded with zeros).
// ---------------------------------------------------------------------------

typedef __attribute__((ext_vector_type(16))) _Float16 v16h;
typedef __attribute__((ext_vector_type(8)))  float    v8f;

#define LSEQ     320      // N * i_p * i_p = 5*64
#define NF       5
#define MIDF     2        // N // 2
#define HH       256
#define NQB      20       // 320 / 16 query blocks
#define NTHREADS 640      // 20 waves
#define NPATCH   1024     // (H/8)^2

// 16-bit A-fragment (16x32) K-index for half h (0..15) given lane group.
// lanes 0-15 : h=0..7 -> K=h,     h=8..15 -> K=h+8   (K16..23)
// lanes 16-31: h=0..7 -> K=h+8,   h=8..15 -> K=h+16  (K24..31)
__device__ __forceinline__ int a_kk(int h, int laneHi) {
  int kk = (h < 8) ? h : (h + 8);
  return kk + (laneHi ? 8 : 0);
}

__global__ __launch_bounds__(NTHREADS)
void dip_attn_kernel(const float* __restrict__ inputs,
                     const float* __restrict__ outputs,
                     const float* __restrict__ Wq, const float* __restrict__ Wk,
                     const float* __restrict__ bq, const float* __restrict__ bk,
                     const float* __restrict__ bv,
                     float* __restrict__ rec_image,          // d_out + 1
                     float* __restrict__ ws_mid,
                     float* __restrict__ ws_all,
                     float* __restrict__ ws_rec)
{
  __shared__ _Float16 qf[LSEQ * 4];          // q, f16, K padded to 4
  __shared__ _Float16 kf[LSEQ * 4];          // k
  __shared__ _Float16 vf[LSEQ * 4];          // v = X + bv
  __shared__ _Float16 Pbuf[NQB * 16 * 32];   // per-wave exp(score) scratch
  __shared__ float    out_lds[LSEQ * 3];     // attention output, f32
  __shared__ float    red[3 * NTHREADS];     // loss reduction scratch

  const int r   = blockIdx.x;
  const int tid = threadIdx.x;
  const int py0 = (r >> 5) * 8;              // patch origin in the 256x256 image
  const int px0 = (r & 31) * 8;

  // ---------------- Phase 1: q/k/v projections into LDS (f16) --------------
  if (tid < LSEQ) {
    const int n  = tid >> 6;                 // frame
    const int pp = tid & 63;                 // ph*8+pw
    const int ph = pp >> 3, pw = pp & 7;
    const int gy = py0 + ph, gx = px0 + pw;
    float xv[3];
    #pragma unroll
    for (int c = 0; c < 3; ++c)
      xv[c] = inputs[((n * 3 + c) * HH + gy) * HH + gx];
    const float rs3 = 0.57735026918962576f; // 1/sqrt(head_dim=3)
    #pragma unroll
    for (int c = 0; c < 3; ++c) {
      float qv = (Wq[c*3+0]*xv[0] + Wq[c*3+1]*xv[1] + Wq[c*3+2]*xv[2] + bq[c]) * rs3;
      float kv =  Wk[c*3+0]*xv[0] + Wk[c*3+1]*xv[1] + Wk[c*3+2]*xv[2] + bk[c];
      float vv =  xv[c] + bv[c];
      qf[tid*4 + c] = (_Float16)qv;
      kf[tid*4 + c] = (_Float16)kv;
      vf[tid*4 + c] = (_Float16)vv;
    }
    qf[tid*4+3] = (_Float16)0.f; kf[tid*4+3] = (_Float16)0.f; vf[tid*4+3] = (_Float16)0.f;
  }
  __syncthreads();

  // ---------------- Phase 2: flash attention, one wave per query block -----
  const int lane   = tid & 31;
  const int wave   = tid >> 5;               // query block id, 0..19
  const int laneHi = lane >> 4;
  const int ln     = lane & 15;
  _Float16* Pw = &Pbuf[wave * 512];

  // A fragment of q (16x32, K=0..2 real, rest zero-padded).
  v16h aQ = {};
  if (!laneHi) {
    #pragma unroll
    for (int h = 0; h < 3; ++h)              // a_kk(h,0)=h for h<8
      aQ[h] = qf[(wave * 16 + ln) * 4 + h];
  }

  float mrun[8], lrun[8];                    // per-row softmax running stats,
  #pragma unroll                             // replicated across the row's 16 lanes
  for (int j = 0; j < 8; ++j) { mrun[j] = -3.0e38f; lrun[j] = 0.f; }
  v8f o = {};                                // f32 O accumulator (cols 0..2 live)

  for (int kc = 0; kc < 10; ++kc) {          // 10 chunks of 32 keys
    const int kb = kc * 32;

    // B fragments of k^T for key tiles [kb, kb+16) and [kb+16, kb+32).
    // B layout: lanes 0-15 hold K=0..7,16..23 (halves h<8 -> K=h); lanes 16-31 K=8..15,24..31,
    // all >= 3 -> zero there.
    v16h b0 = {}, b1 = {};
    if (!laneHi) {
      #pragma unroll
      for (int h = 0; h < 3; ++h) {
        b0[h] = kf[(kb + ln) * 4 + h];
        b1[h] = kf[(kb + 16 + ln) * 4 + h];
      }
    }
    v8f z = {};
    v8f s0 = __builtin_amdgcn_wmma_f32_16x16x32_f16(false, aQ, false, b0, (short)0, z, false, false);
    v8f s1 = __builtin_amdgcn_wmma_f32_16x16x32_f16(false, aQ, false, b1, (short)0, z, false, false);

    // Online softmax over these 32 columns. D layout: element j of lane =
    // row (j + 8*laneHi), col ln.
    #pragma unroll
    for (int j = 0; j < 8; ++j) {
      float t = fmaxf(s0[j], s1[j]);
      #pragma unroll
      for (int m = 1; m < 16; m <<= 1) t = fmaxf(t, __shfl_xor(t, m, 32));
      const float nm    = fmaxf(mrun[j], t);
      const float alpha = __expf(mrun[j] - nm);
      const float p0 = __expf(s0[j] - nm);
      const float p1 = __expf(s1[j] - nm);
      float rssum = p0 + p1;
      #pragma unroll
      for (int m = 1; m < 16; m <<= 1) rssum += __shfl_xor(rssum, m, 32);
      lrun[j] = lrun[j] * alpha + rssum;
      mrun[j] = nm;
      o[j] *= alpha;                          // rescale accumulator row
      const int row = j + 8 * laneHi;
      Pw[row * 32 + ln]      = (_Float16)p0;  // stage P tile row-major in LDS
      Pw[row * 32 + 16 + ln] = (_Float16)p1;
    }
    // LDS ops from one wave are in order: safe to reload Pw immediately.

    // Rebuild P as an A fragment (16x32).
    v16h aP;
    #pragma unroll
    for (int h = 0; h < 16; ++h)
      aP[h] = Pw[ln * 32 + h + 16 * laneHi];

    // V as a B fragment (32x16): rows = keys kb..kb+31, cols = channels (3 live).
    v16h bV = {};
    if (ln < 3) {
      #pragma unroll
      for (int h = 0; h < 16; ++h)
        bV[h] = vf[(kb + h + 16 * laneHi) * 4 + ln];
    }
    o = __builtin_amdgcn_wmma_f32_16x16x32_f16(false, aP, false, bV, (short)0, o, false, false);
  }

  // Normalize and publish this query block's output rows.
  #pragma unroll
  for (int j = 0; j < 8; ++j) {
    if (ln < 3) {
      const int row = j + 8 * laneHi;
      out_lds[(wave * 16 + row) * 3 + ln] = o[j] / lrun[j];
    }
  }
  __syncthreads();

  // ---------------- Phase 3: losses + rec_image ----------------------------
  float mid_s = 0.f, all_s = 0.f, rec_s = 0.f;

  // mid / all losses over (N=5, C=3, 8, 8) = 960 elements.
  for (int idx = tid; idx < NF * 3 * 64; idx += NTHREADS) {
    const int n   = idx / 192;
    const int rem = idx - n * 192;
    const int c   = rem >> 6;
    const int pp  = rem & 63;
    const int ph  = pp >> 3, pw = pp & 7;
    const float a  = out_lds[(n * 64 + pp) * 3 + c];
    const float yv = outputs[((n    * 3 + c) * HH + py0 + ph) * HH + px0 + pw];
    const float ym = outputs[((MIDF * 3 + c) * HH + py0 + ph) * HH + px0 + pw];
    mid_s += (a - ym) * (a - ym);
    all_s += (a - yv) * (a - yv);
  }
  // rec image (mean over frames) + rec loss over (3, 8, 8) = 192 elements.
  for (int idx = tid; idx < 3 * 64; idx += NTHREADS) {
    const int c  = idx >> 6;
    const int pp = idx & 63;
    const int ph = pp >> 3, pw = pp & 7;
    float acc = 0.f;
    #pragma unroll
    for (int n = 0; n < NF; ++n) acc += out_lds[(n * 64 + pp) * 3 + c];
    const float rec = acc * 0.2f;
    rec_image[(c * HH + py0 + ph) * HH + px0 + pw] = rec;
    const float ym = outputs[((MIDF * 3 + c) * HH + py0 + ph) * HH + px0 + pw];
    rec_s += (ym - rec) * (ym - rec);
  }

  // Fixed-order tree reduction of the three partial sums.
  red[tid] = mid_s; red[NTHREADS + tid] = all_s; red[2 * NTHREADS + tid] = rec_s;
  __syncthreads();
  for (int s = 512; s > 0; s >>= 1) {
    if (tid < s && tid + s < NTHREADS) {
      red[tid]               += red[tid + s];
      red[NTHREADS + tid]    += red[NTHREADS + tid + s];
      red[2*NTHREADS + tid]  += red[2*NTHREADS + tid + s];
    }
    __syncthreads();
  }
  if (tid == 0) {
    ws_mid[r] = red[0]           * (1.f / 960.f);            // per-patch mean
    ws_all[r] = red[NTHREADS]    * (1.f / 960.f);
    ws_rec[r] = red[2*NTHREADS]  * (1.f / (3.f * HH * HH));  // image-mean share
  }
}

// Deterministic final fold: loss = sum(mid) + coeff*sum(all) + sum(rec).
__global__ __launch_bounds__(256)
void reduce_loss_kernel(const float* __restrict__ ws,
                        const float* __restrict__ loss_diff,
                        const int* __restrict__ step,
                        const int* __restrict__ max_steps,
                        float* __restrict__ out)
{
  __shared__ float sm[256], sa[256], sr[256];
  const int t = threadIdx.x;
  float m = 0.f, a = 0.f, rr = 0.f;
  for (int i = t; i < NPATCH; i += 256) {
    m  += ws[i];
    a  += ws[NPATCH + i];
    rr += ws[2 * NPATCH + i];
  }
  sm[t] = m; sa[t] = a; sr[t] = rr;
  __syncthreads();
  for (int s = 128; s > 0; s >>= 1) {
    if (t < s) { sm[t] += sm[t+s]; sa[t] += sa[t+s]; sr[t] += sr[t+s]; }
    __syncthreads();
  }
  if (t == 0) {
    const float x  = 1.0f - (float)step[0] / (float)max_steps[0];
    const float x2 = x * x, x4 = x2 * x2, x8 = x4 * x4;
    const float coeff = loss_diff[0] * x8 * x2;   // loss_diff * x^10
    out[0] = sm[0] + coeff * sa[0] + sr[0];
  }
}

extern "C" void kernel_launch(void* const* d_in, const int* in_sizes, int n_in,
                              void* d_out, int out_size, void* d_ws, size_t ws_size,
                              hipStream_t stream) {
  (void)in_sizes; (void)n_in; (void)out_size; (void)ws_size;
  const float* inputs    = (const float*)d_in[0];
  const float* outputs   = (const float*)d_in[1];
  const float* Wq        = (const float*)d_in[2];
  const float* Wk        = (const float*)d_in[3];
  const float* bq        = (const float*)d_in[4];
  const float* bk        = (const float*)d_in[5];
  const float* bv        = (const float*)d_in[6];
  const float* loss_diff = (const float*)d_in[7];
  const int*   step      = (const int*)d_in[8];
  const int*   max_steps = (const int*)d_in[9];

  float* out = (float*)d_out;   // out[0] = loss, out[1..] = rec_image (3,256,256)
  float* ws  = (float*)d_ws;    // 3 * 1024 f32 partials

  dip_attn_kernel<<<NPATCH, NTHREADS, 0, stream>>>(
      inputs, outputs, Wq, Wk, bq, bk, bv,
      out + 1, ws, ws + NPATCH, ws + 2 * NPATCH);
  reduce_loss_kernel<<<1, 256, 0, stream>>>(ws, loss_diff, step, max_steps, out);
}
